// EdgeAwareSAGEConv_7902739825245
// MI455X (gfx1250) — matrix operand order repaired and checked by reading, not compiled, not run on hardware
//
#include <hip/hip_runtime.h>
#include <hip/hip_bf16.h>

// CDNA5 WMMA types
typedef __attribute__((ext_vector_type(16))) _Float16 v16h;
typedef __attribute__((ext_vector_type(8)))  float    v8f;

#define N_SRC    50000
#define N_DSTN   50000
#define N_EDGE   1000000
#define IN_C     64
#define OUT_C    256
#define EDGE_DIM 32

// ---------------------------------------------------------------------------
// Kernel 1: ew[e] = mean_n relu( edge_attr[e,:] @ W_edge[:,n] + b_edge[n] )
// Each wave owns FOUR 16-edge tiles (4 A fragments + 4 accumulators). The
// n-tile loop is kept ROLLED (#pragma unroll 1) so the B fragment is a fresh
// ds_load each iteration (round-1 showed full unroll spills B to scratch).
// Each B fragment feeds 4 back-to-back independent v_wmma; the relu/acc VALU
// dual-issues in their shadow (verified in round-2 asm).
// ---------------------------------------------------------------------------
__global__ void ew_wmma_kernel(const float* __restrict__ edge_attr,
                               const float* __restrict__ W_edge,
                               const float* __restrict__ b_edge,
                               float* __restrict__ ew) {
    __shared__ __align__(32) _Float16 sW[OUT_C * EDGE_DIM];   // [n][k] 16 KB
    __shared__ float sB[OUT_C];

    const int tid = threadIdx.x;
    for (int i = tid; i < OUT_C * EDGE_DIM; i += blockDim.x) {
        int n = i >> 5, k = i & 31;                            // transpose to n-major
        sW[i] = (_Float16)W_edge[k * OUT_C + n];
    }
    for (int i = tid; i < OUT_C; i += blockDim.x) sB[i] = b_edge[i];
    __syncthreads();

    const int lane   = tid & 31;
    const int half   = lane >> 4;          // 0: lanes 0-15, 1: lanes 16-31
    const int l15    = lane & 15;
    const int kbaseA = half * 16;          // A-frag K offset (contiguous, matches B)
    const int koffB  = half * 16;          // B-frag K offset within LDS row

    const int wavesPerBlock = blockDim.x >> 5;
    const int waveId = blockIdx.x * wavesPerBlock + (tid >> 5);
    const int nWaves = gridDim.x * wavesPerBlock;
    const int nGroups = N_EDGE / (16 * 4);  // 4 tiles per wave, exactly 15625

    for (int g = waveId; g < nGroups; g += nWaves) {
        const int tile0 = g * 4;

        // Prefetch the next group's A rows (streaming 128MB of edge_attr is
        // the bandwidth-critical path of this kernel).
        {
            int gn = g + nWaves;
            if (gn < nGroups) {
                const float* pf = edge_attr + (size_t)(gn * 64 + l15) * EDGE_DIM;
                __builtin_prefetch(pf, 0, 1);
            }
        }

        // Load 4 A fragments: 4 x (16 edges x 32 K), f32 -> f16 on the fly.
        v16h a[4];
#pragma unroll
        for (int t = 0; t < 4; ++t) {
            const float* arow =
                edge_attr + (size_t)((tile0 + t) * 16 + l15) * EDGE_DIM;
#pragma unroll
            for (int i = 0; i < 8; ++i) a[t][i]     = (_Float16)arow[kbaseA + i];
#pragma unroll
            for (int i = 0; i < 8; ++i) a[t][8 + i] = (_Float16)arow[kbaseA + 8 + i];
        }

        v8f acc[4] = {};
#pragma unroll 1
        for (int nt = 0; nt < OUT_C / 16; ++nt) {
            const int n = nt * 16 + l15;
            const v16h b = *(const v16h*)&sW[n * EDGE_DIM + koffB];
            const float bias = sB[n];

            v8f c[4];
            const v8f z = {};
#pragma unroll
            for (int t = 0; t < 4; ++t)
                c[t] = __builtin_amdgcn_wmma_f32_16x16x32_f16(
                           false, a[t], false, b, (short)0, z, false, false);
#pragma unroll
            for (int t = 0; t < 4; ++t) {
#pragma unroll
                for (int r = 0; r < 8; ++r) {
                    float v = c[t][r] + bias;
                    acc[t][r] += (v > 0.0f) ? v : 0.0f;   // relu, sum over n
                }
            }
        }

        // Column reduction: each half-wave's 16 lanes hold 16 columns of the
        // same 8 rows -> butterfly over masks 1,2,4,8 (stays within half).
#pragma unroll
        for (int m = 1; m < 16; m <<= 1) {
#pragma unroll
            for (int t = 0; t < 4; ++t)
#pragma unroll
                for (int r = 0; r < 8; ++r)
                    acc[t][r] += __shfl_xor(acc[t][r], m, 32);
        }
        if (l15 == 0) {
#pragma unroll
            for (int t = 0; t < 4; ++t)
#pragma unroll
                for (int r = 0; r < 8; ++r)
                    ew[(tile0 + t) * 16 + half * 8 + r] =
                        acc[t][r] * (1.0f / OUT_C);
        }
    }
}

// ---------------------------------------------------------------------------
// Kernel 2: zero int counters (must run every launch; graph replays do not
// re-poison the workspace).
// ---------------------------------------------------------------------------
__global__ void zero_i32_kernel(int* __restrict__ p, int n) {
    int i = blockIdx.x * blockDim.x + threadIdx.x;
    if (i < n) p[i] = 0;
}

// ---------------------------------------------------------------------------
// Kernel 3a: histogram of destinations (1M int atomics vs 64M fp32 atomics
// of the naive scatter).
// ---------------------------------------------------------------------------
__global__ void count_kernel(const int* __restrict__ dst, int* __restrict__ cnt_i) {
    int e = blockIdx.x * blockDim.x + threadIdx.x;
    if (e >= N_EDGE) return;
    atomicAdd(&cnt_i[dst[e]], 1);
}

// ---------------------------------------------------------------------------
// Kernel 3b: single-block exclusive scan over the 50000 counts -> CSR row
// offsets (and a second "cursor" copy consumed by the fill pass).
// 1024 threads x 49 serial elements + Hillis-Steele LDS scan of partials.
// ---------------------------------------------------------------------------
__global__ void scan_kernel(const int* __restrict__ cnt_i,
                            int* __restrict__ offs,
                            int* __restrict__ cursor) {
    __shared__ int part[1024];
    const int tid = threadIdx.x;
    const int CH = (N_DSTN + 1023) / 1024;     // 49
    const int base = tid * CH;

    int s = 0;
    for (int i = 0; i < CH; ++i) {
        int idx = base + i;
        if (idx < N_DSTN) s += cnt_i[idx];
    }
    part[tid] = s;
    __syncthreads();
    for (int d = 1; d < 1024; d <<= 1) {
        int v = 0;
        if (tid >= d) v = part[tid - d];
        __syncthreads();
        if (tid >= d) part[tid] += v;
        __syncthreads();
    }
    int run = (tid == 0) ? 0 : part[tid - 1];  // exclusive base for this chunk
    for (int i = 0; i < CH; ++i) {
        int idx = base + i;
        if (idx < N_DSTN) {
            offs[idx]   = run;
            cursor[idx] = run;
            run += cnt_i[idx];
        }
    }
}

// ---------------------------------------------------------------------------
// Kernel 3c: bucket-fill edge ids by destination (CSR column array).
// ---------------------------------------------------------------------------
__global__ void fill_kernel(const int* __restrict__ dst,
                            int* __restrict__ cursor,
                            int* __restrict__ eidx) {
    int e = blockIdx.x * blockDim.x + threadIdx.x;
    if (e >= N_EDGE) return;
    int pos = atomicAdd(&cursor[dst[e]], 1);
    eidx[pos] = e;
}

// ---------------------------------------------------------------------------
// Kernel 3d: contention-free gather-mean. One wave per destination node; each
// lane owns 2 of the 64 channels. Edge id / weight / src index are wave-
// uniform (scalar loads); x_src rows (12.8 MB), ew (4 MB), src (4 MB) and
// eidx (4 MB) are all L2-resident on the 192 MB L2, so this runs at L2
// bandwidth instead of atomic-RMW throughput. Writes agg already divided by
// max(deg,1).
// ---------------------------------------------------------------------------
__global__ void gather_kernel(const float* __restrict__ x_src,
                              const int* __restrict__ src,
                              const float* __restrict__ ew,
                              const int* __restrict__ offs,
                              const int* __restrict__ cnt_i,
                              const int* __restrict__ eidx,
                              float* __restrict__ agg) {
    const int wavesPerBlock = blockDim.x >> 5;
    const int d = blockIdx.x * wavesPerBlock + (threadIdx.x >> 5);
    if (d >= N_DSTN) return;
    const int lane = threadIdx.x & 31;

    const int start = offs[d];
    const int deg   = cnt_i[d];
    const float2* xs = (const float2*)x_src;

    float ax = 0.0f, ay = 0.0f;
    for (int j = 0; j < deg; ++j) {
        const int   e = eidx[start + j];   // wave-uniform
        const float w = ew[e];             // wave-uniform
        const int   s = src[e];            // wave-uniform
        const float2 v = xs[(size_t)s * 32 + lane];
        ax += v.x * w;
        ay += v.y * w;
    }
    const float invd = 1.0f / (deg > 1 ? (float)deg : 1.0f);
    float2 o; o.x = ax * invd; o.y = ay * invd;
    ((float2*)agg)[(size_t)d * 32 + lane] = o;
}

// ---------------------------------------------------------------------------
// Kernel 4: out = agg @ W_l + b_l + x_dst @ W_r
// Fused as [agg | x_dst] (N x 128) @ [W_l ; W_r] (128 x 256): K=128 -> four
// accumulating v_wmma_f32_16x16x32_f16 per 16x16 output tile. The n-tile loop
// is kept rolled so B fragments stream from LDS (no register spills).
// ---------------------------------------------------------------------------
__global__ void sage_out_kernel(const float* __restrict__ agg,
                                const float* __restrict__ x_dst,
                                const float* __restrict__ W_l,
                                const float* __restrict__ b_l,
                                const float* __restrict__ W_r,
                                float* __restrict__ out) {
    __shared__ __align__(32) _Float16 sW[OUT_C * 128];   // [n][k] 64 KB
    __shared__ float sB[OUT_C];

    const int tid = threadIdx.x;
    for (int i = tid; i < OUT_C * 128; i += blockDim.x) {
        int n = i >> 7, k = i & 127;
        float w = (k < 64) ? W_l[k * OUT_C + n] : W_r[(k - 64) * OUT_C + n];
        sW[i] = (_Float16)w;
    }
    for (int i = tid; i < OUT_C; i += blockDim.x) sB[i] = b_l[i];
    __syncthreads();

    const int lane   = tid & 31;
    const int half   = lane >> 4;
    const int l15    = lane & 15;
    const int kbaseA = half * 16;
    const int koffB  = half * 16;

    const int wavesPerBlock = blockDim.x >> 5;
    const int waveId = blockIdx.x * wavesPerBlock + (tid >> 5);
    const int nWaves = gridDim.x * wavesPerBlock;
    const int nTiles = N_DSTN / 16;        // exactly 3125

    for (int tile = waveId; tile < nTiles; tile += nWaves) {
        const int row = tile * 16 + l15;
        const float* arow = agg   + (size_t)row * IN_C;
        const float* xrow = x_dst + (size_t)row * IN_C;

        v16h afrag[4];                      // K = 128 as four 16x32 A fragments
#pragma unroll
        for (int kc = 0; kc < 4; ++kc) {
#pragma unroll
            for (int i = 0; i < 16; ++i) {
                int kk = kc * 32 + kbaseA + i;
                float v = (kk < 64) ? arow[kk] : xrow[kk - 64];
                afrag[kc][i] = (_Float16)v;
            }
        }

#pragma unroll 1
        for (int nt = 0; nt < OUT_C / 16; ++nt) {
            const int n = nt * 16 + l15;
            v8f c = {};
#pragma unroll
            for (int kc = 0; kc < 4; ++kc) {
                const v16h b = *(const v16h*)&sW[n * 128 + kc * 32 + koffB];
                c = __builtin_amdgcn_wmma_f32_16x16x32_f16(
                        false, afrag[kc], false, b, (short)0, c, false, false);
            }
            const float bias = sB[n];
            float* orow = out + (size_t)(tile * 16 + half * 8) * OUT_C + n;
#pragma unroll
            for (int r = 0; r < 8; ++r)
                orow[(size_t)r * OUT_C] = c[r] + bias;
        }
    }
}

// ---------------------------------------------------------------------------
// Launcher. Inputs (setup_inputs order):
//  0 x_src (50000x64 f32)   1 x_dst (50000x64 f32)  2 edge_index (2xE int)
//  3 edge_attr (Ex32 f32)   4 W_edge (32x256 f32)   5 b_edge (256 f32)
//  6 W_l (64x256 f32)       7 b_l (256 f32)         8 W_r (64x256 f32)
// Output: (50000x256 f32).
// Workspace: ew | agg | eidx | cnt_i | offs | cursor  (~21.5 MB).
// ---------------------------------------------------------------------------
extern "C" void kernel_launch(void* const* d_in, const int* in_sizes, int n_in,
                              void* d_out, int out_size, void* d_ws, size_t ws_size,
                              hipStream_t stream) {
    const float* x_src     = (const float*)d_in[0];
    const float* x_dst     = (const float*)d_in[1];
    const int*   edge_idx  = (const int*)d_in[2];
    const float* edge_attr = (const float*)d_in[3];
    const float* W_edge    = (const float*)d_in[4];
    const float* b_edge    = (const float*)d_in[5];
    const float* W_l       = (const float*)d_in[6];
    const float* b_l       = (const float*)d_in[7];
    const float* W_r       = (const float*)d_in[8];
    float*       out       = (float*)d_out;

    const int* src = edge_idx;
    const int* dst = edge_idx + N_EDGE;

    float* ew     = (float*)d_ws;                        // N_EDGE f32
    float* agg    = ew + N_EDGE;                         // N_DSTN*IN_C f32
    int*   eidx   = (int*)(agg + (size_t)N_DSTN * IN_C); // N_EDGE i32
    int*   cnt_i  = eidx + N_EDGE;                       // N_DSTN i32
    int*   offs   = cnt_i + N_DSTN;                      // N_DSTN i32
    int*   cursor = offs + N_DSTN;                       // N_DSTN i32

    // 1) zero the destination histogram (only buffer that accumulates).
    zero_i32_kernel<<<(N_DSTN + 255) / 256, 256, 0, stream>>>(cnt_i, N_DSTN);

    // 2) edge weights (WMMA). 512 blocks x 8 waves over 15625 4-tile groups.
    ew_wmma_kernel<<<512, 256, 0, stream>>>(edge_attr, W_edge, b_edge, ew);

    // 3) CSR build: count -> scan -> fill (2M atomics total).
    count_kernel<<<(N_EDGE + 255) / 256, 256, 0, stream>>>(dst, cnt_i);
    scan_kernel<<<1, 1024, 0, stream>>>(cnt_i, offs, cursor);
    fill_kernel<<<(N_EDGE + 255) / 256, 256, 0, stream>>>(dst, cursor, eidx);

    // 4) contention-free gather-mean: one wave per destination node.
    gather_kernel<<<(N_DSTN + 7) / 8, 256, 0, stream>>>(x_src, src, ew,
                                                        offs, cnt_i, eidx, agg);

    // 5) fused SAGE output GEMM (WMMA). One wave per 16-row tile.
    sage_out_kernel<<<(N_DSTN / 16 + 7) / 8, 256, 0, stream>>>(agg, x_dst,
                                                               W_l, b_l, W_r, out);
}
